// SimpleVectorQuantizer_70248485093769
// MI455X (gfx1250) — compile-verified, matrix operand order checked by reading, not run on hardware
//
#include <hip/hip_runtime.h>

// Problem constants (from reference)
#define BB   32
#define DD   96
#define TT   8192
#define KK   256
#define NPTS (BB * TT)            // 262144 points
#define BDT  (BB * DD * TT)       // 25165824 quant elements
#define BETA 0.25f

#define BLOCK_THREADS  128
#define WAVES_PER_BLK  4
#define PTS_PER_WAVE   16
#define PTS_PER_BLK    (WAVES_PER_BLK * PTS_PER_WAVE)   // 64
#define LDS_STRIDE     98   // padded f16 row stride: 196 B (dword aligned, conflict-free)

typedef __attribute__((ext_vector_type(16))) _Float16 v16h;
typedef __attribute__((ext_vector_type(8)))  float    v8f;

union Frag { v16h h; unsigned int u[8]; };

// Build a 16-bit A/B WMMA fragment from an LDS row per the CDNA5 layout:
// lane L holds row (L%16); half = (L/16)*8.
//   VGPR v in 0..3 -> K = ks + half + 2v, +1
//   VGPR v in 4..7 -> K = ks + half + 16 + 2(v-4), +1
// (ks + half) is even, rows are 196B apart -> all dword-aligned loads.
__device__ __forceinline__ v16h load_frag(const _Float16* rowp, int half, int ks) {
    Frag f;
    const unsigned int* p = (const unsigned int*)(rowp + ks + half);
    f.u[0] = p[0]; f.u[1] = p[1]; f.u[2] = p[2]; f.u[3] = p[3];
    const unsigned int* q = (const unsigned int*)(rowp + ks + half + 16);
    f.u[4] = q[0]; f.u[5] = q[1]; f.u[6] = q[2]; f.u[7] = q[3];
    return f.h;
}

__global__ void vq_init_kernel(float* acc) {
    acc[0] = 0.0f;
}

__global__ __launch_bounds__(BLOCK_THREADS)
void vq_main_kernel(const float* __restrict__ x,
                    const float* __restrict__ cb,
                    float* __restrict__ out,
                    float* __restrict__ loss_acc) {
    __shared__ __align__(16) _Float16 s_cb[KK][LDS_STRIDE];                      // 50176 B
    __shared__ __align__(16) _Float16 s_x[WAVES_PER_BLK][PTS_PER_WAVE][LDS_STRIDE]; // 12544 B
    __shared__ float s_cn[KK];                                                   // 1024 B
    __shared__ float s_xn[WAVES_PER_BLK][PTS_PER_WAVE];                          // 256 B
    __shared__ int   s_idx[WAVES_PER_BLK][PTS_PER_WAVE];                         // 256 B

    const int tid  = threadIdx.x;
    const int wave = tid >> 5;
    const int lane = tid & 31;

    // ---- Stage codebook (f32 -> f16) + f16-consistent squared norms ----
    for (int r = tid; r < KK; r += BLOCK_THREADS) {
        const float* src = cb + r * DD;
        float nrm = 0.0f;
        #pragma unroll 8
        for (int d = 0; d < DD; ++d) {
            _Float16 h = (_Float16)src[d];
            s_cb[r][d] = h;
            float hv = (float)h;
            nrm += hv * hv;
        }
        s_cn[r] = nrm;
    }

    // ---- Stage this wave's 16 points of x (f32 -> f16) ----
    const long n0   = (long)blockIdx.x * PTS_PER_BLK + (long)wave * PTS_PER_WAVE;
    const int  bb0  = (int)(n0 / TT);
    const int  t0   = (int)(n0 % TT);     // tile never crosses a batch boundary (TT % 16 == 0)
    const size_t xbase = (size_t)bb0 * DD * TT + (size_t)t0;

    for (int i = lane; i < PTS_PER_WAVE * DD; i += 32) {
        int m = i & 15;       // point within tile (t offset) -> coalesced per half-wave
        int d = i >> 4;       // dimension
        float v = x[xbase + (size_t)d * TT + (size_t)m];
        s_x[wave][m][d] = (_Float16)v;
    }
    __syncthreads();

    // ---- Per-point squared norms from the f16 tile ----
    if (tid < PTS_PER_BLK) {
        int w = tid >> 4, m = tid & 15;
        float s = 0.0f;
        #pragma unroll 8
        for (int d = 0; d < DD; ++d) {
            float v = (float)s_x[w][m][d];
            s += v * v;
        }
        s_xn[w][m] = s;
    }
    __syncthreads();

    // ---- WMMA distance GEMM + running argmin ----
    const int half  = (lane >> 4) << 3;   // 0 or 8
    const int nlane = lane & 15;          // N column within tile / M row for A

    const _Float16* arow = &s_x[wave][nlane][0];
    v16h a0 = load_frag(arow, half, 0);
    v16h a1 = load_frag(arow, half, 32);
    v16h a2 = load_frag(arow, half, 64);

    float xn[8];
    #pragma unroll
    for (int r = 0; r < 8; ++r) xn[r] = s_xn[wave][r + half];

    float best[8];
    int   bi[8];
    #pragma unroll
    for (int r = 0; r < 8; ++r) { best[r] = 3.402823466e38f; bi[r] = 0; }

    for (int kb = 0; kb < KK / 16; ++kb) {
        const _Float16* brow = &s_cb[kb * 16 + nlane][0];
        v8f acc = {};
        acc = __builtin_amdgcn_wmma_f32_16x16x32_f16(false, a0, false,
                  load_frag(brow, half, 0),  (short)0, acc, false, false);
        acc = __builtin_amdgcn_wmma_f32_16x16x32_f16(false, a1, false,
                  load_frag(brow, half, 32), (short)0, acc, false, false);
        acc = __builtin_amdgcn_wmma_f32_16x16x32_f16(false, a2, false,
                  load_frag(brow, half, 64), (short)0, acc, false, false);

        const int   nidx = kb * 16 + nlane;
        const float cn   = s_cn[nidx];
        #pragma unroll
        for (int r = 0; r < 8; ++r) {
            float dist = xn[r] + cn - 2.0f * acc[r];
            if (dist < best[r]) { best[r] = dist; bi[r] = nidx; }
        }
    }

    // ---- Argmin reduction across the 16 N-lanes of each half-wave ----
    #pragma unroll
    for (int r = 0; r < 8; ++r) {
        #pragma unroll
        for (int off = 1; off < 16; off <<= 1) {
            float od = __shfl_xor(best[r], off, 16);
            int   oi = __shfl_xor(bi[r],   off, 16);
            if (od < best[r] || (od == best[r] && oi < bi[r])) {
                best[r] = od; bi[r] = oi;
            }
        }
    }
    if (nlane == 0) {
        #pragma unroll
        for (int r = 0; r < 8; ++r) s_idx[wave][half + r] = bi[r];
    }
    __syncthreads();

    // ---- Outputs: indices (as float), exact-f32 quant, loss accumulation ----
    float* idx_out = out + BDT;
    if (lane < 16) idx_out[n0 + lane] = (float)s_idx[wave][lane];

    const int m  = lane & 15;                 // point within tile
    const int ci = s_idx[wave][m];
    const float* crow = cb + (size_t)ci * DD; // exact f32 codebook row (L2-hot)
    const size_t baddr = xbase + (size_t)m;   // + d*TT

    float lsum = 0.0f;
    #pragma unroll 4
    for (int j = 0; j < DD / 2; ++j) {
        int d0 = 2 * j + (lane >> 4);
        float q  = crow[d0];
        size_t a = baddr + (size_t)d0 * TT;   // half-wave writes 16 consecutive floats
        float xv = x[a];
        float df = q - xv;
        lsum += df * df;
        out[a] = q;
    }

    #pragma unroll
    for (int off = 16; off > 0; off >>= 1) lsum += __shfl_xor(lsum, off, 32);
    if (lane == 0) atomicAdd(loss_acc, lsum);
}

__global__ void vq_finalize_kernel(const float* __restrict__ acc,
                                   float* __restrict__ out) {
    float mean = acc[0] / (float)((long)NPTS * DD);
    out[BDT + NPTS]     = mean;          // loss_vq
    out[BDT + NPTS + 1] = BETA * mean;   // loss_commit (same mean, scaled)
}

extern "C" void kernel_launch(void* const* d_in, const int* in_sizes, int n_in,
                              void* d_out, int out_size, void* d_ws, size_t ws_size,
                              hipStream_t stream) {
    const float* x  = (const float*)d_in[0];   // [B, D, T] f32
    const float* cb = (const float*)d_in[1];   // [K, D] f32
    float* out = (float*)d_out;                // quant | idx | loss_vq | loss_commit
    float* acc = (float*)d_ws;                 // scalar loss accumulator

    vq_init_kernel<<<1, 1, 0, stream>>>(acc);
    vq_main_kernel<<<NPTS / PTS_PER_BLK, BLOCK_THREADS, 0, stream>>>(x, cb, out, acc);
    vq_finalize_kernel<<<1, 1, 0, stream>>>(acc, out);
}
